// MultiHeadAttention_59236188946632
// MI455X (gfx1250) — matrix operand order repaired and checked by reading, not compiled, not run on hardware
//
#include <hip/hip_runtime.h>

// ---------------------------------------------------------------------------
// MI455X (gfx1250, wave32) multi-head attention forward.
// All matmuls via v_wmma_f32_16x16x32_bf16. Shared operands (weights in the
// GEMMs, K/V tiles in attention) are staged once per block into LDS through a
// double-buffered GLOBAL_LOAD_ASYNC_TO_LDS_B128 pipeline (ASYNCcnt), removing
// the 8x per-wave redundant global traffic that would otherwise bottleneck the
// WGP memory path long before the matrix cores saturate.
// ---------------------------------------------------------------------------

typedef __attribute__((ext_vector_type(16))) __bf16 v16bf;
typedef __attribute__((ext_vector_type(8)))  __bf16 v8bf;
typedef __attribute__((ext_vector_type(8)))  float  v8f;
typedef __attribute__((ext_vector_type(4)))  int    v4i;

#define DMODEL 1024
#define SEQ    2048
#define NHEAD  16
#define HDIM   64
#define BATCH  2
#define MROWS  4096   // BATCH*SEQ

enum { OUT_BF16 = 0, OUT_VT = 1, OUT_F32 = 2 };

// ---- async global->LDS copy path (guarded; falls back to reg staging) -----
#if defined(__HIP_DEVICE_COMPILE__) && \
    __has_builtin(__builtin_amdgcn_global_load_async_to_lds_b128) && \
    __has_builtin(__builtin_amdgcn_s_wait_asynccnt)
#define HAVE_ASYNC_LDS 1
#else
#define HAVE_ASYNC_LDS 0
#endif

#if HAVE_ASYNC_LDS
// builtin signature (per hipcc diagnostic): param0 = v4i in AS1 (__device__),
// param1 = v4i in AS3 (__shared__), then imm offset + imm cpol.
typedef __attribute__((address_space(1))) v4i as1_v4i;
typedef __attribute__((address_space(3))) v4i as3_v4i;
#define ASYNC_WAIT(N) __builtin_amdgcn_s_wait_asynccnt(N)
#else
#define ASYNC_WAIT(N) ((void)0)
#endif

// copy 16 bytes (8 bf16) global -> LDS
__device__ __forceinline__ void cp16_g2l(__bf16* l, const __bf16* g) {
#if HAVE_ASYNC_LDS
    // generic LDS pointer low 32 bits == wave-relative LDS offset (aperture rule)
    __builtin_amdgcn_global_load_async_to_lds_b128(
        (as1_v4i*)(uintptr_t)g, (as3_v4i*)(uint32_t)(uintptr_t)l, 0, 0);
#else
    *(v8bf*)l = *(const v8bf*)g;
#endif
}

// ---------------------------------------------------------------------------
// elementwise f32 -> bf16
// ---------------------------------------------------------------------------
__global__ __launch_bounds__(256) void mha_cvt_bf16(const float* __restrict__ in,
                                                    __bf16* __restrict__ out, int n) {
    int i = blockIdx.x * 256 + threadIdx.x;
    if (i < n) out[i] = (__bf16)in[i];
}

// ---------------------------------------------------------------------------
// W[k][n] f32 -> Wt[n][k] bf16  (16x16 LDS tile transpose)
// ---------------------------------------------------------------------------
__global__ __launch_bounds__(256) void mha_transpose_w(const float* __restrict__ W,
                                                       __bf16* __restrict__ Wt) {
    __shared__ float tile[16][17];
    const int tx = threadIdx.x & 15;
    const int ty = threadIdx.x >> 4;
    const int k0 = blockIdx.x * 16;
    const int n0 = blockIdx.y * 16;
    tile[ty][tx] = W[(size_t)(k0 + ty) * DMODEL + n0 + tx];
    __syncthreads();
    Wt[(size_t)(n0 + ty) * DMODEL + k0 + tx] = (__bf16)tile[tx][ty];
}

// ---------------------------------------------------------------------------
// GEMM: Out[m][n] = sum_k A[m][k] * Wt[n][k] + bias[n]
// Block = 256 threads (8 waves), tile 256(M) x 64(N). Wave: 32(M) x 64(N),
// 8 accumulators, K-step 32 -> 8 WMMAs per step. The shared Wt tile
// (64 x 32 bf16 = 4 KB) is double-buffered in LDS via async copies.
// ---------------------------------------------------------------------------
__global__ __launch_bounds__(256) void mha_gemm_bf16(const __bf16* __restrict__ A,
                                                     const __bf16* __restrict__ Wt,
                                                     const float* __restrict__ bias,
                                                     void* __restrict__ Out, int mode) {
    __shared__ alignas(64) __bf16 wtile[2][64 * 32];   // 2 x 4 KB

    const int tid  = threadIdx.x;
    const int lane = tid & 31;
    const int wave = tid >> 5;
    const int hl   = lane >> 4;
    const int l16  = lane & 15;
    const int m0   = blockIdx.y * 256 + wave * 32;
    const int n0   = blockIdx.x * 64;

    // staging: thread -> (row n_local, 16B chunk); 256 threads cover 4 KB
    const int srow   = tid >> 2;          // 0..63
    const int schunk = (tid & 3) * 8;     // bf16 offset within row
    const __bf16* wsrc = Wt + (size_t)(n0 + srow) * DMODEL + schunk;

    v8f acc[2][4] = {};
    const __bf16* Arow0 = A + (size_t)(m0 + l16) * DMODEL;
    const __bf16* Arow1 = Arow0 + 16 * DMODEL;

    cp16_g2l(&wtile[0][srow * 32 + schunk], wsrc);    // prefetch k=0

    int buf = 0;
    for (int k = 0; k < DMODEL; k += 32, buf ^= 1) {
        if (k + 32 < DMODEL) {
            cp16_g2l(&wtile[buf ^ 1][srow * 32 + schunk], wsrc + k + 32);
            ASYNC_WAIT(1);
        } else {
            ASYNC_WAIT(0);
        }
        __syncthreads();

        // A fragments 16x32 (per-wave rows, straight from global; no sharing)
        union { v16bf v; v8bf p[2]; } af[2];
        af[0].p[0] = *(const v8bf*)(Arow0 + k + 16 * hl);
        af[0].p[1] = *(const v8bf*)(Arow0 + k + 16 * hl + 8);
        af[1].p[0] = *(const v8bf*)(Arow1 + k + 16 * hl);
        af[1].p[1] = *(const v8bf*)(Arow1 + k + 16 * hl + 8);
#pragma unroll
        for (int j = 0; j < 4; ++j) {
            v16bf bf = *(const v16bf*)&wtile[buf][(16 * j + l16) * 32 + 16 * hl];
            acc[0][j] = __builtin_amdgcn_wmma_f32_16x16x32_bf16(
                false, af[0].v, false, bf, (short)0, acc[0][j], false, false);
            acc[1][j] = __builtin_amdgcn_wmma_f32_16x16x32_bf16(
                false, af[1].v, false, bf, (short)0, acc[1][j], false, false);
        }
        __syncthreads();
    }

#pragma unroll
    for (int mi = 0; mi < 2; ++mi) {
#pragma unroll
        for (int j = 0; j < 4; ++j) {
            const int n = n0 + 16 * j + l16;
            const float bj = bias[n];
#pragma unroll
            for (int r = 0; r < 8; ++r) {
                const int m = m0 + 16 * mi + r + 8 * hl;   // C layout row
                const float val = acc[mi][j][r] + bj;
                if (mode == OUT_BF16) {
                    ((__bf16*)Out)[(size_t)m * DMODEL + n] = (__bf16)val;
                } else if (mode == OUT_VT) {
                    const int bb = m >> 11;
                    const int ss = m & (SEQ - 1);
                    ((__bf16*)Out)[((size_t)(bb * DMODEL + n)) * SEQ + ss] = (__bf16)val;
                } else {
                    ((float*)Out)[(size_t)m * DMODEL + n] = val;
                }
            }
        }
    }
}

// ---------------------------------------------------------------------------
// Flash attention. Grid = (SEQ/128, NHEAD, BATCH), block = 256 (8 waves).
// All 8 waves share (b,h) and the key-tile walk, so K (32x64) and V (64x32)
// tiles are staged once per block in double-buffered LDS via async copies.
// Each wave: 16 query rows x head dim 64; per 32-key tile: 4 WMMAs for
// S = Q K^T, online softmax (shfl_xor row reductions in 16-lane halves),
// P f32->bf16 via per-wave LDS relayout, 4 WMMAs for O += P V.
// ---------------------------------------------------------------------------
__global__ __launch_bounds__(256) void mha_flash_attn(const __bf16* __restrict__ Q,
                                                      const __bf16* __restrict__ K,
                                                      const __bf16* __restrict__ Vt,
                                                      __bf16* __restrict__ AO) {
    __shared__ alignas(64) __bf16 ktile[2][32 * 64];   // 2 x 4 KB: [key][d]
    __shared__ alignas(64) __bf16 vtile[2][64 * 32];   // 2 x 4 KB: [d][key]
    __shared__ alignas(64) __bf16 plds[8][16 * 32];    // per-wave P tile

    const int tid  = threadIdx.x;
    const int lane = tid & 31;
    const int wave = tid >> 5;
    const int hl   = lane >> 4;
    const int l16  = lane & 15;

    const int b  = blockIdx.z;
    const int h  = blockIdx.y;
    const int q0 = blockIdx.x * 128 + wave * 16;
    const size_t rowbase = (size_t)b * SEQ;

    // staging maps (256 threads cover each 4 KB tile with one 16B chunk each)
    const int krow = tid >> 3;               // 0..31  key index
    const int kchk = (tid & 7) * 8;          // d offset
    const int vrow = tid >> 2;               // 0..63  d index
    const int vchk = (tid & 3) * 8;          // key offset
    const __bf16* ksrc = K + (rowbase + krow) * DMODEL + h * HDIM + kchk;
    const __bf16* vsrc = Vt + ((size_t)(b * DMODEL + h * HDIM + vrow)) * SEQ + vchk;

    // Q A-fragments: 16 x 64 head slice = two 16x32 fragments
    const __bf16* qrow = Q + (rowbase + q0 + l16) * DMODEL + h * HDIM;
    union { v16bf v; v8bf p[2]; } qa[2];
#pragma unroll
    for (int t = 0; t < 2; ++t) {
        qa[t].p[0] = *(const v8bf*)(qrow + 32 * t + 16 * hl);
        qa[t].p[1] = *(const v8bf*)(qrow + 32 * t + 16 * hl + 8);
    }

    v8f o[4] = {};
    float mrow[8], lrow[8];
#pragma unroll
    for (int r = 0; r < 8; ++r) { mrow[r] = -3.0e38f; lrow[r] = 0.0f; }

    const float LOG2E = 1.44269504088896f;
    const float scale = 0.125f;   // 1/sqrt(64)

    // prefetch key-tile 0
    cp16_g2l(&ktile[0][krow * 64 + kchk], ksrc);
    cp16_g2l(&vtile[0][vrow * 32 + vchk], vsrc);

    int buf = 0;
    for (int k0 = 0; k0 < SEQ; k0 += 32, buf ^= 1) {
        if (k0 + 32 < SEQ) {
            cp16_g2l(&ktile[buf ^ 1][krow * 64 + kchk], ksrc + (size_t)(k0 + 32) * DMODEL);
            cp16_g2l(&vtile[buf ^ 1][vrow * 32 + vchk], vsrc + (k0 + 32));
            ASYNC_WAIT(2);
        } else {
            ASYNC_WAIT(0);
        }
        __syncthreads();

        // ---- scores: two 16x16 C tiles (key cols 0..15 / 16..31)
        v8f sc[2] = {};
#pragma unroll
        for (int c = 0; c < 2; ++c) {
#pragma unroll
            for (int t = 0; t < 2; ++t) {
                v16bf kb = *(const v16bf*)&ktile[buf][(16 * c + l16) * 64 + 32 * t + 16 * hl];
                sc[c] = __builtin_amdgcn_wmma_f32_16x16x32_bf16(
                    false, qa[t].v, false, kb, (short)0, sc[c], false, false);
            }
        }
        // ---- online softmax (row m = r + 8*hl lives in one 16-lane half)
#pragma unroll
        for (int r = 0; r < 8; ++r) {
            float v0 = sc[0][r] * scale;
            float v1 = sc[1][r] * scale;
            float tmax = fmaxf(v0, v1);
            tmax = fmaxf(tmax, __shfl_xor(tmax, 1, 32));
            tmax = fmaxf(tmax, __shfl_xor(tmax, 2, 32));
            tmax = fmaxf(tmax, __shfl_xor(tmax, 4, 32));
            tmax = fmaxf(tmax, __shfl_xor(tmax, 8, 32));
            const float mnew  = fmaxf(mrow[r], tmax);
            const float alpha = __builtin_exp2f((mrow[r] - mnew) * LOG2E);
            const float p0 = __builtin_exp2f((v0 - mnew) * LOG2E);
            const float p1 = __builtin_exp2f((v1 - mnew) * LOG2E);
            float psum = p0 + p1;
            psum += __shfl_xor(psum, 1, 32);
            psum += __shfl_xor(psum, 2, 32);
            psum += __shfl_xor(psum, 4, 32);
            psum += __shfl_xor(psum, 8, 32);
            lrow[r] = lrow[r] * alpha + psum;
            mrow[r] = mnew;
#pragma unroll
            for (int j = 0; j < 4; ++j) o[j][r] *= alpha;
            // C layout (m = r + 8*hl, n = l16) -> row-major P tile in LDS
            plds[wave][(r + 8 * hl) * 32 + l16]      = (__bf16)p0;
            plds[wave][(r + 8 * hl) * 32 + 16 + l16] = (__bf16)p1;
        }
        // ---- P as A fragment (same-wave LDS ops are in-order)
        union { v16bf v; v8bf p[2]; } pa;
        pa.p[0] = *(const v8bf*)&plds[wave][l16 * 32 + 16 * hl];
        pa.p[1] = *(const v8bf*)&plds[wave][l16 * 32 + 16 * hl + 8];
        // ---- O += P x V
#pragma unroll
        for (int j = 0; j < 4; ++j) {
            v16bf vb = *(const v16bf*)&vtile[buf][(16 * j + l16) * 32 + 16 * hl];
            o[j] = __builtin_amdgcn_wmma_f32_16x16x32_bf16(
                false, pa.v, false, vb, (short)0, o[j], false, false);
        }
        __syncthreads();
    }

    // finalize + store AO (bf16 row-major, ready for the Wo GEMM)
#pragma unroll
    for (int r = 0; r < 8; ++r) {
        const float inv = 1.0f / lrow[r];
        const size_t m = rowbase + q0 + r + 8 * hl;
#pragma unroll
        for (int j = 0; j < 4; ++j) {
            AO[m * DMODEL + h * HDIM + 16 * j + l16] = (__bf16)(o[j][r] * inv);
        }
    }
}

// ---------------------------------------------------------------------------
extern "C" void kernel_launch(void* const* d_in, const int* in_sizes, int n_in,
                              void* d_out, int out_size, void* d_ws, size_t ws_size,
                              hipStream_t stream) {
    const float* x  = (const float*)d_in[0];
    const float* Wq = (const float*)d_in[1];
    const float* bq = (const float*)d_in[2];
    const float* Wk = (const float*)d_in[3];
    const float* bk = (const float*)d_in[4];
    const float* Wv = (const float*)d_in[5];
    const float* bv = (const float*)d_in[6];
    const float* Wo = (const float*)d_in[7];
    const float* bo = (const float*)d_in[8];

    char* ws = (char*)d_ws;
    size_t off = 0;
    const size_t SZ_X = (size_t)MROWS * DMODEL * sizeof(__bf16);   // 8 MB
    const size_t SZ_W = (size_t)DMODEL * DMODEL * sizeof(__bf16);  // 2 MB

    __bf16* xb  = (__bf16*)(ws + off); off += SZ_X;
    __bf16* Wqt = (__bf16*)(ws + off); off += SZ_W;
    __bf16* Wkt = (__bf16*)(ws + off); off += SZ_W;
    __bf16* Wvt = (__bf16*)(ws + off); off += SZ_W;
    __bf16* Wot = (__bf16*)(ws + off); off += SZ_W;
    __bf16* Qb  = (__bf16*)(ws + off); off += SZ_X;
    __bf16* Kb  = (__bf16*)(ws + off); off += SZ_X;
    __bf16* VtT = (__bf16*)(ws + off); off += SZ_X;
    __bf16* AO  = (__bf16*)(ws + off); off += SZ_X;

    dim3 blk(256);

    // 1. precision conversion + weight transposes
    mha_cvt_bf16<<<(MROWS * DMODEL) / 256, blk, 0, stream>>>(x, xb, MROWS * DMODEL);
    dim3 tg(DMODEL / 16, DMODEL / 16);
    mha_transpose_w<<<tg, blk, 0, stream>>>(Wq, Wqt);
    mha_transpose_w<<<tg, blk, 0, stream>>>(Wk, Wkt);
    mha_transpose_w<<<tg, blk, 0, stream>>>(Wv, Wvt);
    mha_transpose_w<<<tg, blk, 0, stream>>>(Wo, Wot);

    // 2. Q/K/V projections (V stored transposed as [B, D, S])
    dim3 gg(DMODEL / 64, MROWS / 256);
    mha_gemm_bf16<<<gg, blk, 0, stream>>>(xb, Wqt, bq, Qb,  OUT_BF16);
    mha_gemm_bf16<<<gg, blk, 0, stream>>>(xb, Wkt, bk, Kb,  OUT_BF16);
    mha_gemm_bf16<<<gg, blk, 0, stream>>>(xb, Wvt, bv, VtT, OUT_VT);

    // 3. flash attention
    dim3 ag(SEQ / 128, NHEAD, BATCH);
    mha_flash_attn<<<ag, blk, 0, stream>>>(Qb, Kb, VtT, AO);

    // 4. output projection -> fp32 d_out
    mha_gemm_bf16<<<gg, blk, 0, stream>>>(AO, Wot, bo, d_out, OUT_F32);
}
